// Prototype_IP_56779467653590
// MI455X (gfx1250) — matrix operand order repaired and checked by reading, not compiled, not run on hardware
//
#include <hip/hip_runtime.h>
#include <hip/hip_bf16.h>

typedef __bf16 bf16;
typedef __attribute__((ext_vector_type(16))) __bf16 v16bf;
typedef __attribute__((ext_vector_type(8)))  float  v8f;

#define C_DIM  2048
#define K_DIM  200
#define KPAD   208    // 13 tiles of 16 (GEMM1 N)
#define KPAD2  224    // 7 k-steps of 32 (GEMM2 K)
#define NC     1000
#define NCPAD  1008   // 63 tiles of 16 (GEMM2 N)
#define B_ROWS 16384

__device__ __forceinline__ float wave_sum(float v) {
#pragma unroll
  for (int off = 16; off > 0; off >>= 1) v += __shfl_xor(v, off, 32);
  return v;
}
__device__ __forceinline__ float wave_max(float v) {
#pragma unroll
  for (int off = 16; off > 0; off >>= 1) v = fmaxf(v, __shfl_xor(v, off, 32));
  return v;
}

// ---------------- Kernel 1a: L2-normalize concepts -> bf16 proto[KPAD][C_DIM]
__global__ void proto_norm_kernel(const float* __restrict__ concepts,
                                  bf16* __restrict__ proto) {
  const int k = blockIdx.x;
  const int tid = threadIdx.x;
  __shared__ float red[256];
  if (k >= K_DIM) {  // zero pad rows 200..207
    for (int c = tid; c < C_DIM; c += 256) proto[(size_t)k * C_DIM + c] = (bf16)0.0f;
    return;
  }
  float s = 0.f;
  for (int c = tid; c < C_DIM; c += 256) {
    float v = concepts[(size_t)k * C_DIM + c];
    s += v * v;
  }
  red[tid] = s;
  __syncthreads();
  for (int st = 128; st > 0; st >>= 1) {
    if (tid < st) red[tid] += red[tid + st];
    __syncthreads();
  }
  const float scale = 1.0f / fmaxf(sqrtf(red[0]), 1e-12f);
  for (int c = tid; c < C_DIM; c += 256)
    proto[(size_t)k * C_DIM + c] = (bf16)(concepts[(size_t)k * C_DIM + c] * scale);
}

// ---------------- Kernel 1b: fc_w f32[NC][K_DIM] -> bf16 fcw[NCPAD][KPAD2] (zero pad)
__global__ void fcw_cvt_kernel(const float* __restrict__ fcw,
                               bf16* __restrict__ fcwb) {
  const int n = blockIdx.x;
  const int tid = threadIdx.x;
  if (tid < KPAD2) {
    bf16 v = (bf16)0.0f;
    if (n < NC && tid < K_DIM) v = (bf16)fcw[(size_t)n * K_DIM + tid];
    fcwb[(size_t)n * KPAD2 + tid] = v;
  }
}

// ---------------- Kernel 2: fused GEMM1 (x @ protoT) + LayerNorm + top-k mask -> bf16 cf
__global__ __launch_bounds__(128) void gemm1_ln_topk_kernel(
    const float* __restrict__ x, const bf16* __restrict__ proto,
    const float* __restrict__ lnw, const float* __restrict__ lnb,
    const int* __restrict__ feat_idx_p, bf16* __restrict__ cf) {
  __shared__ float lds_cf[4][16][KPAD];  // 53,248 B

  const int tid = threadIdx.x;
  const int w  = tid >> 5;      // wave 0..3
  const int l  = tid & 31;      // lane
  const int r0 = blockIdx.x * 64 + w * 16;
  const int m  = l & 15;
  const int hi = (l >> 4) & 1;

  const v8f vzero = {0.f, 0.f, 0.f, 0.f, 0.f, 0.f, 0.f, 0.f};
  v8f acc[13];
#pragma unroll
  for (int t = 0; t < 13; ++t) acc[t] = vzero;

  const float* xrow = x + (size_t)(r0 + m) * C_DIM;

  for (int c0 = 0; c0 < C_DIM; c0 += 32) {
    // A fragment: 16-bit A 16x32 layout. lanes0-15: K{0-7,16-23}; lanes16-31: K{8-15,24-31}
    const float* pa = xrow + c0 + hi * 8;
    __builtin_prefetch(pa + 64, 0, 0);  // global_prefetch_b8 (speculative, next k-steps)
    float4 q0 = *(const float4*)(pa);
    float4 q1 = *(const float4*)(pa + 4);
    float4 q2 = *(const float4*)(pa + 16);
    float4 q3 = *(const float4*)(pa + 20);
    v16bf a;
    a[0]  = (bf16)q0.x; a[1]  = (bf16)q0.y; a[2]  = (bf16)q0.z; a[3]  = (bf16)q0.w;
    a[4]  = (bf16)q1.x; a[5]  = (bf16)q1.y; a[6]  = (bf16)q1.z; a[7]  = (bf16)q1.w;
    a[8]  = (bf16)q2.x; a[9]  = (bf16)q2.y; a[10] = (bf16)q2.z; a[11] = (bf16)q2.w;
    a[12] = (bf16)q3.x; a[13] = (bf16)q3.y; a[14] = (bf16)q3.z; a[15] = (bf16)q3.w;

    const int kb = c0 + hi * 8;  // B 32x16 layout: lane=N, lanes0-15 K{0-7,16-23}, lanes16-31 K{8-15,24-31}
#pragma unroll
    for (int t = 0; t < 13; ++t) {
      const int n = t * 16 + m;
      union { uint4 u[2]; v16bf v; } bb;
      const uint4* pb = (const uint4*)(proto + (size_t)n * C_DIM + kb);
      bb.u[0] = pb[0];
      bb.u[1] = pb[2];
      acc[t] = __builtin_amdgcn_wmma_f32_16x16x32_bf16(false, a, false, bb.v,
                                                       (short)0, acc[t], false, false);
    }
  }

  // Spill C/D tiles to LDS: VGPR j, lane l -> row (j + hi*8), col (t*16 + m)
#pragma unroll
  for (int t = 0; t < 13; ++t) {
#pragma unroll
    for (int j = 0; j < 8; ++j) {
      lds_cf[w][j + hi * 8][t * 16 + m] = acc[t][j];
    }
  }
  __syncthreads();

  const int iters = K_DIM - *feat_idx_p;  // 20: threshold = 20th-largest |value|

  for (int row = 0; row < 16; ++row) {
    float v[7], y[7], a0[7], aw[7];
    float s = 0.f, s2 = 0.f;
#pragma unroll
    for (int i = 0; i < 7; ++i) {
      const int col = l + 32 * i;
      const float val = (col < K_DIM) ? lds_cf[w][row][col] : 0.f;
      v[i] = val;
      s += val;
      s2 += val * val;
    }
    s  = wave_sum(s);
    s2 = wave_sum(s2);
    const float mean = s * (1.0f / K_DIM);
    const float var  = s2 * (1.0f / K_DIM) - mean * mean;
    const float rs   = rsqrtf(var + 1e-5f);
#pragma unroll
    for (int i = 0; i < 7; ++i) {
      const int col = l + 32 * i;
      if (col < K_DIM) {
        const float yy = (v[i] - mean) * rs * lnw[col] + lnb[col];
        y[i]  = yy;
        a0[i] = fabsf(yy);
        aw[i] = a0[i];
      } else {
        y[i]  = 0.f;
        a0[i] = -1.f;
        aw[i] = -1.f;
      }
    }
    // iteratively extract 'iters' maxima; threshold = last extracted (exact order statistic)
    float thres = 0.f;
    for (int it = 0; it < iters; ++it) {
      float lm = aw[0];
      int ls = 0;
#pragma unroll
      for (int i = 1; i < 7; ++i)
        if (aw[i] > lm) { lm = aw[i]; ls = i; }
      const float mx = wave_max(lm);
      const unsigned long long msk = __ballot(lm == mx);
      const int first = __ffsll((long long)msk) - 1;
      if (l == first) aw[ls] = -2.f;  // remove exactly one instance
      thres = mx;
    }
    // strict-> mask (matches |feat| - thres > 0), write bf16, zero K pad 200..223
    const size_t rg = (size_t)(r0 + row) * KPAD2;
#pragma unroll
    for (int i = 0; i < 7; ++i) {
      const int col = l + 32 * i;  // covers 0..223 exactly
      bf16 o = (bf16)0.0f;
      if (col < K_DIM && a0[i] > thres) o = (bf16)y[i];
      cf[rg + col] = o;
    }
  }
}

// ---------------- Kernel 3: GEMM2  out = cf @ fcwT + b
__global__ __launch_bounds__(128) void gemm2_kernel(
    const bf16* __restrict__ cf, const bf16* __restrict__ fcwb,
    const float* __restrict__ fcb, float* __restrict__ out) {
  const int tid = threadIdx.x;
  const int w  = tid >> 5;
  const int l  = tid & 31;
  const int r0 = blockIdx.x * 64 + w * 16;
  const int nb = blockIdx.y * 112;
  const int m  = l & 15;
  const int hi = (l >> 4) & 1;

  const v8f vzero = {0.f, 0.f, 0.f, 0.f, 0.f, 0.f, 0.f, 0.f};
  v8f acc[7];
#pragma unroll
  for (int t = 0; t < 7; ++t) acc[t] = vzero;

  const bf16* arow = cf + (size_t)(r0 + m) * KPAD2;

#pragma unroll
  for (int c0 = 0; c0 < KPAD2; c0 += 32) {
    union { uint4 u[2]; v16bf v; } aa;
    const bf16* pa = arow + c0 + hi * 8;
    aa.u[0] = *(const uint4*)(pa);
    aa.u[1] = *(const uint4*)(pa + 16);
    const int kb = c0 + hi * 8;
#pragma unroll
    for (int t = 0; t < 7; ++t) {
      union { uint4 u[2]; v16bf v; } bb;
      const uint4* pb = (const uint4*)(fcwb + (size_t)(nb + t * 16 + m) * KPAD2 + kb);
      bb.u[0] = pb[0];
      bb.u[1] = pb[2];
      acc[t] = __builtin_amdgcn_wmma_f32_16x16x32_bf16(false, aa.v, false, bb.v,
                                                       (short)0, acc[t], false, false);
    }
  }

#pragma unroll
  for (int t = 0; t < 7; ++t) {
    const int col = nb + t * 16 + m;
    if (col < NC) {
      const float bias = fcb[col];
#pragma unroll
      for (int j = 0; j < 8; ++j) {
        const int row = r0 + j + hi * 8;
        out[(size_t)row * NC + col] = acc[t][j] + bias;
      }
    }
  }
}

extern "C" void kernel_launch(void* const* d_in, const int* in_sizes, int n_in,
                              void* d_out, int out_size, void* d_ws, size_t ws_size,
                              hipStream_t stream) {
  const float* x        = (const float*)d_in[0];
  const float* concepts = (const float*)d_in[1];
  const float* lnw      = (const float*)d_in[2];
  const float* lnb      = (const float*)d_in[3];
  const float* fcw      = (const float*)d_in[4];
  const float* fcb      = (const float*)d_in[5];
  const int*   fi       = (const int*)d_in[6];
  float* out = (float*)d_out;

  char* ws = (char*)d_ws;
  bf16* proto = (bf16*)ws;                                   // 208*2048*2  = 851,968 B
  bf16* fcwb  = (bf16*)(ws + 851968);                        // 1008*224*2 = 451,584 B
  bf16* cf    = (bf16*)(ws + 851968 + 451584);               // 16384*224*2 = 7,340,032 B

  proto_norm_kernel<<<KPAD, 256, 0, stream>>>(concepts, proto);
  fcw_cvt_kernel<<<NCPAD, 256, 0, stream>>>(fcw, fcwb);
  gemm1_ln_topk_kernel<<<B_ROWS / 64, 128, 0, stream>>>(x, proto, lnw, lnb, fi, cf);
  gemm2_kernel<<<dim3(B_ROWS / 64, NCPAD / 112), 128, 0, stream>>>(cf, fcwb, fcb, out);
}
